// MoELayer_76605036692010
// MI455X (gfx1250) — matrix operand order repaired and checked by reading, not compiled, not run on hardware
//
#include <hip/hip_runtime.h>

#define N_TOK 4096
#define DIM   1024
#define HID   4096
#define NEXP  8
#define TOPK  2
#define NSLOT (N_TOK * TOPK)   // 8192

typedef __bf16 bf16_t;
typedef __attribute__((ext_vector_type(16))) __bf16 v16bf;
typedef __attribute__((ext_vector_type(8)))  __bf16 v8bf;
typedef __attribute__((ext_vector_type(4)))  __bf16 v4bf;
typedef __attribute__((ext_vector_type(8)))  float  v8f;
typedef unsigned int u32x4 __attribute__((ext_vector_type(4)));
typedef int          i32x4 __attribute__((ext_vector_type(4)));
typedef int          i32x8 __attribute__((ext_vector_type(8)));

// ---------------------------------------------------------------------------
// TDM: 2D tile load (rows x rowElems bf16 elements, row stride in elements)
// into LDS at lds_addr. D# layout per CDNA5 ISA.
// ---------------------------------------------------------------------------
__device__ __forceinline__ void tdm_load_2d(unsigned lds_addr, const void* gptr,
                                            unsigned rows, unsigned rowElems,
                                            unsigned tensorRowLen,
                                            unsigned long long strideElems) {
    unsigned long long ga = (unsigned long long)(uintptr_t)gptr;
    u32x4 g0;
    g0.x = 1u;                                                   // count=1, user mode
    g0.y = lds_addr;                                             // lds_addr[31:0]
    g0.z = (unsigned)(ga & 0xFFFFFFFFu);                         // global_addr lo
    g0.w = (unsigned)((ga >> 32) & 0x01FFFFFFu) | (2u << 30);    // addr hi | type=2
    i32x8 g1;
    g1[0] = (int)(1u << 16);                                     // data_size=1 (2B)
    g1[1] = (int)((tensorRowLen & 0xFFFFu) << 16);               // tensor_dim0 lo16
    g1[2] = (int)(((tensorRowLen >> 16) & 0xFFFFu) |
                  ((rows & 0xFFFFu) << 16));                     // dim0 hi | dim1 lo
    g1[3] = (int)(((rows >> 16) & 0xFFFFu) |
                  ((rowElems & 0xFFFFu) << 16));                 // dim1 hi | tile_dim0
    g1[4] = (int)(rows & 0xFFFFu);                               // tile_dim1 (tile_dim2=0)
    g1[5] = (int)(unsigned)(strideElems & 0xFFFFFFFFu);          // dim0_stride lo32
    g1[6] = (int)(unsigned)((strideElems >> 32) & 0xFFFFu);      // dim0_stride hi16
    g1[7] = 0;                                                   // dim1_stride hi (unused)
    i32x4 z4 = {};
#if __clang_major__ >= 23
    i32x8 z8 = {};
    __builtin_amdgcn_tensor_load_to_lds(g0, g1, z4, z4, z8, 0);
#else
    __builtin_amdgcn_tensor_load_to_lds(g0, g1, z4, z4, 0);
#endif
}

// ---------------------------------------------------------------------------
// 0. zero per-expert counters
// ---------------------------------------------------------------------------
__global__ void moe_zero_cnt(int* cnt) {
    if (threadIdx.x < NEXP) cnt[threadIdx.x] = 0;
}

// ---------------------------------------------------------------------------
// 1. fp32 -> bf16 convert (x)
// ---------------------------------------------------------------------------
__global__ void cvt_x(const float* __restrict__ x, bf16_t* __restrict__ xb) {
    int i = (blockIdx.x * blockDim.x + threadIdx.x) * 4;
    if (i < N_TOK * DIM) {
        float4 v = *(const float4*)(x + i);
        xb[i + 0] = (bf16_t)v.x;
        xb[i + 1] = (bf16_t)v.y;
        xb[i + 2] = (bf16_t)v.z;
        xb[i + 3] = (bf16_t)v.w;
    }
}

// ---------------------------------------------------------------------------
// 2. per-expert transpose + convert: src [E][R][C] f32 -> dst [E][C][R] bf16
// ---------------------------------------------------------------------------
template <int R, int C>
__global__ void cvt_transpose(const float* __restrict__ src, bf16_t* __restrict__ dst) {
    __shared__ float tile[32][33];
    int e  = blockIdx.z;
    int r0 = blockIdx.x * 32;
    int c0 = blockIdx.y * 32;
    const float* s = src + (size_t)e * R * C;
    bf16_t*      d = dst + (size_t)e * R * C;
    int tx = threadIdx.x & 31;
    int ty = threadIdx.x >> 5;
    #pragma unroll
    for (int i = ty; i < 32; i += 8)
        tile[i][tx] = s[(size_t)(r0 + i) * C + (c0 + tx)];
    __syncthreads();
    #pragma unroll
    for (int i = ty; i < 32; i += 8)
        d[(size_t)(c0 + i) * R + (r0 + tx)] = (bf16_t)tile[tx][i];
}

// ---------------------------------------------------------------------------
// 3. gating: scores, top-2, softmax, compaction (one wave per token)
// ---------------------------------------------------------------------------
__global__ void moe_gate(const float* __restrict__ x, const float* __restrict__ Wg,
                         const float* __restrict__ bg,
                         float* __restrict__ gw, int* __restrict__ perm,
                         int* __restrict__ cnt) {
    int tok  = blockIdx.x * (blockDim.x >> 5) + (threadIdx.x >> 5);
    int lane = threadIdx.x & 31;
    if (tok >= N_TOK) return;

    const float* xr = x + (size_t)tok * DIM;
    float acc[NEXP];
    #pragma unroll
    for (int e = 0; e < NEXP; ++e) acc[e] = 0.f;

    for (int d = lane; d < DIM; d += 32) {
        float xv = xr[d];
        const float* wr = Wg + (size_t)d * NEXP;
        #pragma unroll
        for (int e = 0; e < NEXP; ++e) acc[e] += xv * wr[e];
    }
    #pragma unroll
    for (int off = 16; off > 0; off >>= 1) {
        #pragma unroll
        for (int e = 0; e < NEXP; ++e)
            acc[e] += __shfl_xor(acc[e], off, 32);
    }
    if (lane == 0) {
        #pragma unroll
        for (int e = 0; e < NEXP; ++e) acc[e] += bg[e];
        int i0 = 0;
        #pragma unroll
        for (int e = 1; e < NEXP; ++e) if (acc[e] > acc[i0]) i0 = e;
        int i1 = (i0 == 0) ? 1 : 0;
        #pragma unroll
        for (int e = 0; e < NEXP; ++e)
            if (e != i0 && acc[e] > acc[i1]) i1 = e;
        float m  = acc[i0];
        float e0 = __expf(acc[i0] - m);
        float e1 = __expf(acc[i1] - m);
        float inv = 1.f / (e0 + e1);
        int slot0 = tok * TOPK, slot1 = slot0 + 1;
        gw[slot0] = e0 * inv;
        gw[slot1] = e1 * inv;
        int p0 = atomicAdd(&cnt[i0], 1);
        perm[i0 * N_TOK + p0] = slot0;
        int p1 = atomicAdd(&cnt[i1], 1);
        perm[i1 * N_TOK + p1] = slot1;
    }
}

// ---------------------------------------------------------------------------
// LDS fragment helpers (contiguous k order k = 16*hi + s in both A and B)
// ---------------------------------------------------------------------------
__device__ __forceinline__ v16bf cat8(v8bf lo, v8bf up) {
    return __builtin_shufflevector(lo, up, 0, 1, 2, 3, 4, 5, 6, 7,
                                          8, 9, 10, 11, 12, 13, 14, 15);
}
// A 16x32 from LDS tile [row][32]: lane l holds row M=l&15, hi=lane>>4 selects
// 16 contiguous K at hi*16.
__device__ __forceinline__ v16bf frag_a(const bf16_t* tile, int row, int hi) {
    const bf16_t* p = tile + row * 32 + hi * 16;
    return cat8(*(const v8bf*)p, *(const v8bf*)(p + 8));
}
// B 32x16 from LDS tile [col][32]: lane holds 16 contiguous K at hi*16.
__device__ __forceinline__ v16bf frag_b(const bf16_t* tile, int col, int hi) {
    const bf16_t* p = tile + col * 32 + hi * 16;
    return cat8(*(const v8bf*)p, *(const v8bf*)(p + 8));
}

// ---------------------------------------------------------------------------
// 4. FC1: h[slot] = relu(x[tok] @ W1[e] + b1[e])
//    block 256 thr (8 waves): tile M=32 x N=256; TDM double-buffered B in LDS
// ---------------------------------------------------------------------------
__global__ void moe_fc1(const bf16_t* __restrict__ xb, const bf16_t* __restrict__ W1t,
                        const float* __restrict__ b1,
                        const int* __restrict__ perm, const int* __restrict__ cnt,
                        bf16_t* __restrict__ hbuf) {
    int e     = blockIdx.z;
    int mTile = blockIdx.x;
    int count = cnt[e];
    if (mTile * 32 >= count) return;

    __shared__ __align__(16) bf16_t sB[2][256 * 32];   // [col][k]
    __shared__ __align__(16) bf16_t sA[2][32 * 32];    // [row][k]
    __shared__ int rowTok[32];
    __shared__ int rowSlot[32];

    int tid = threadIdx.x;
    if (tid < 32) {
        int pos  = mTile * 32 + tid;
        int slot = perm[e * N_TOK + ((pos < count) ? pos : mTile * 32)];
        rowSlot[tid] = slot;
        rowTok[tid]  = slot >> 1;
    }
    __syncthreads();

    int nBase = blockIdx.y * 256;
    const bf16_t* bSrc = W1t + ((size_t)e * HID + nBase) * DIM;   // rows=cols of out
    // cooperative A staging: thread -> (row, 4 elems)
    int arow = tid >> 3;
    int acol = (tid & 7) * 4;
    const bf16_t* aRow = xb + (size_t)rowTok[arow] * DIM;

    // prologue: stage k-block 0 into buffer 0
    *(v4bf*)&sA[0][arow * 32 + acol] = *(const v4bf*)(aRow + acol);
    if (tid < 32)
        tdm_load_2d((unsigned)(uintptr_t)&sB[0][0], bSrc, 256, 32, DIM, DIM);

    int wave = tid >> 5;
    int lane = tid & 31;
    int l    = lane & 15;
    int hi   = lane >> 4;
    int nw   = wave * 32;   // wave's 32 columns within block tile

    v8f c00 = {}, c01 = {}, c10 = {}, c11 = {};
    const int NKB = DIM / 32;
    for (int kb = 0; kb < NKB; ++kb) {
        int cur = kb & 1;
        if (tid < 32) __builtin_amdgcn_s_wait_tensorcnt(0);
        __syncthreads();   // buf[cur] ready; buf[1-cur] free (prev compute done)
        if (kb + 1 < NKB) {
            int k0n = (kb + 1) * 32;
            *(v4bf*)&sA[1 - cur][arow * 32 + acol] = *(const v4bf*)(aRow + k0n + acol);
            if (tid < 32)
                tdm_load_2d((unsigned)(uintptr_t)&sB[1 - cur][0],
                            bSrc + k0n, 256, 32, DIM, DIM);
        }
        v16bf a0 = frag_a(&sA[cur][0], l, hi);
        v16bf a1 = frag_a(&sA[cur][0], 16 + l, hi);
        v16bf bfr0 = frag_b(&sB[cur][0], nw + l, hi);
        v16bf bfr1 = frag_b(&sB[cur][0], nw + 16 + l, hi);
        c00 = __builtin_amdgcn_wmma_f32_16x16x32_bf16(false, a0, false, bfr0,
                                                      (short)0, c00, false, false);
        c01 = __builtin_amdgcn_wmma_f32_16x16x32_bf16(false, a0, false, bfr1,
                                                      (short)0, c01, false, false);
        c10 = __builtin_amdgcn_wmma_f32_16x16x32_bf16(false, a1, false, bfr0,
                                                      (short)0, c10, false, false);
        c11 = __builtin_amdgcn_wmma_f32_16x16x32_bf16(false, a1, false, bfr1,
                                                      (short)0, c11, false, false);
    }

    float bias0 = b1[(size_t)e * HID + nBase + nw + l];
    float bias1 = b1[(size_t)e * HID + nBase + nw + 16 + l];
    #pragma unroll
    for (int r = 0; r < 8; ++r) {
        int m0 = r + hi * 8;         // C layout: lanes16-31 hold M=8..15
        int m1 = 16 + m0;
        if (mTile * 32 + m0 < count) {
            bf16_t* orow = hbuf + (size_t)rowSlot[m0] * HID + nBase + nw;
            orow[l]      = (bf16_t)fmaxf(c00[r] + bias0, 0.f);
            orow[16 + l] = (bf16_t)fmaxf(c01[r] + bias1, 0.f);
        }
        if (mTile * 32 + m1 < count) {
            bf16_t* orow = hbuf + (size_t)rowSlot[m1] * HID + nBase + nw;
            orow[l]      = (bf16_t)fmaxf(c10[r] + bias0, 0.f);
            orow[16 + l] = (bf16_t)fmaxf(c11[r] + bias1, 0.f);
        }
    }
}

// ---------------------------------------------------------------------------
// 5. FC2: y[slot] = h[slot] @ W2[e] + b2[e]
// ---------------------------------------------------------------------------
__global__ void moe_fc2(const bf16_t* __restrict__ hbuf, const bf16_t* __restrict__ W2t,
                        const float* __restrict__ b2,
                        const int* __restrict__ perm, const int* __restrict__ cnt,
                        float* __restrict__ y) {
    int e     = blockIdx.z;
    int mTile = blockIdx.x;
    int count = cnt[e];
    if (mTile * 32 >= count) return;

    __shared__ __align__(16) bf16_t sB[2][256 * 32];
    __shared__ __align__(16) bf16_t sA[2][32 * 32];
    __shared__ int rowSlot[32];

    int tid = threadIdx.x;
    if (tid < 32) {
        int pos = mTile * 32 + tid;
        rowSlot[tid] = perm[e * N_TOK + ((pos < count) ? pos : mTile * 32)];
    }
    __syncthreads();

    int nBase = blockIdx.y * 256;
    const bf16_t* bSrc = W2t + ((size_t)e * DIM + nBase) * HID;
    int arow = tid >> 3;
    int acol = (tid & 7) * 4;
    const bf16_t* aRow = hbuf + (size_t)rowSlot[arow] * HID;

    *(v4bf*)&sA[0][arow * 32 + acol] = *(const v4bf*)(aRow + acol);
    if (tid < 32)
        tdm_load_2d((unsigned)(uintptr_t)&sB[0][0], bSrc, 256, 32, HID, HID);

    int wave = tid >> 5;
    int lane = tid & 31;
    int l    = lane & 15;
    int hi   = lane >> 4;
    int nw   = wave * 32;

    v8f c00 = {}, c01 = {}, c10 = {}, c11 = {};
    const int NKB = HID / 32;
    for (int kb = 0; kb < NKB; ++kb) {
        int cur = kb & 1;
        if (tid < 32) __builtin_amdgcn_s_wait_tensorcnt(0);
        __syncthreads();
        if (kb + 1 < NKB) {
            int k0n = (kb + 1) * 32;
            *(v4bf*)&sA[1 - cur][arow * 32 + acol] = *(const v4bf*)(aRow + k0n + acol);
            if (tid < 32)
                tdm_load_2d((unsigned)(uintptr_t)&sB[1 - cur][0],
                            bSrc + k0n, 256, 32, HID, HID);
        }
        v16bf a0 = frag_a(&sA[cur][0], l, hi);
        v16bf a1 = frag_a(&sA[cur][0], 16 + l, hi);
        v16bf bfr0 = frag_b(&sB[cur][0], nw + l, hi);
        v16bf bfr1 = frag_b(&sB[cur][0], nw + 16 + l, hi);
        c00 = __builtin_amdgcn_wmma_f32_16x16x32_bf16(false, a0, false, bfr0,
                                                      (short)0, c00, false, false);
        c01 = __builtin_amdgcn_wmma_f32_16x16x32_bf16(false, a0, false, bfr1,
                                                      (short)0, c01, false, false);
        c10 = __builtin_amdgcn_wmma_f32_16x16x32_bf16(false, a1, false, bfr0,
                                                      (short)0, c10, false, false);
        c11 = __builtin_amdgcn_wmma_f32_16x16x32_bf16(false, a1, false, bfr1,
                                                      (short)0, c11, false, false);
    }

    float bias0 = b2[(size_t)e * DIM + nBase + nw + l];
    float bias1 = b2[(size_t)e * DIM + nBase + nw + 16 + l];
    #pragma unroll
    for (int r = 0; r < 8; ++r) {
        int m0 = r + hi * 8;
        int m1 = 16 + m0;
        if (mTile * 32 + m0 < count) {
            float* orow = y + (size_t)rowSlot[m0] * DIM + nBase + nw;
            orow[l]      = c00[r] + bias0;
            orow[16 + l] = c01[r] + bias1;
        }
        if (mTile * 32 + m1 < count) {
            float* orow = y + (size_t)rowSlot[m1] * DIM + nBase + nw;
            orow[l]      = c10[r] + bias0;
            orow[16 + l] = c11[r] + bias1;
        }
    }
}

// ---------------------------------------------------------------------------
// 6. combine: out[n] = w0 * y[2n] + w1 * y[2n+1]
// ---------------------------------------------------------------------------
__global__ void moe_combine(const float* __restrict__ y, const float* __restrict__ gw,
                            float* __restrict__ out) {
    int i = (blockIdx.x * blockDim.x + threadIdx.x) * 4;
    if (i < N_TOK * DIM) {
        int n = i / DIM;
        int d = i - n * DIM;
        float w0 = gw[n * 2 + 0];
        float w1 = gw[n * 2 + 1];
        float4 a = *(const float4*)(y + (size_t)(n * 2 + 0) * DIM + d);
        float4 b = *(const float4*)(y + (size_t)(n * 2 + 1) * DIM + d);
        float4 o;
        o.x = w0 * a.x + w1 * b.x;
        o.y = w0 * a.y + w1 * b.y;
        o.z = w0 * a.z + w1 * b.z;
        o.w = w0 * a.w + w1 * b.w;
        *(float4*)(out + i) = o;
    }
}

// ---------------------------------------------------------------------------
// launch
// ---------------------------------------------------------------------------
extern "C" void kernel_launch(void* const* d_in, const int* in_sizes, int n_in,
                              void* d_out, int out_size, void* d_ws, size_t ws_size,
                              hipStream_t stream) {
    const float* x  = (const float*)d_in[0];
    const float* Wg = (const float*)d_in[1];
    const float* bg = (const float*)d_in[2];
    const float* W1 = (const float*)d_in[3];
    const float* b1 = (const float*)d_in[4];
    const float* W2 = (const float*)d_in[5];
    const float* b2 = (const float*)d_in[6];
    float* out = (float*)d_out;

    char* ws = (char*)d_ws;
    bf16_t* xb   = (bf16_t*)ws;                            // N*D bf16      (8 MB)
    bf16_t* W1t  = xb   + (size_t)N_TOK * DIM;             // E*H*D bf16    (64 MB)
    bf16_t* W2t  = W1t  + (size_t)NEXP * DIM * HID;        // E*D*H bf16    (64 MB)
    bf16_t* hbuf = W2t  + (size_t)NEXP * DIM * HID;        // NSLOT*H bf16  (64 MB)
    float*  ybuf = (float*)(hbuf + (size_t)NSLOT * HID);   // NSLOT*D f32   (32 MB)
    float*  gw   = ybuf + (size_t)NSLOT * DIM;             // NSLOT f32
    int*    perm = (int*)(gw + NSLOT);                     // E*N int
    int*    cnt  = perm + (size_t)NEXP * N_TOK;            // E int

    moe_zero_cnt<<<1, 32, 0, stream>>>(cnt);

    cvt_x<<<(N_TOK * DIM / 4 + 255) / 256, 256, 0, stream>>>(x, xb);
    cvt_transpose<DIM, HID><<<dim3(DIM / 32, HID / 32, NEXP), 256, 0, stream>>>(W1, W1t);
    cvt_transpose<HID, DIM><<<dim3(HID / 32, DIM / 32, NEXP), 256, 0, stream>>>(W2, W2t);

    moe_gate<<<N_TOK / 8, 256, 0, stream>>>(x, Wg, bg, gw, perm, cnt);

    moe_fc1<<<dim3(N_TOK / 32, HID / 256, NEXP), 256, 0, stream>>>(
        xb, W1t, b1, perm, cnt, hbuf);
    moe_fc2<<<dim3(N_TOK / 32, DIM / 256, NEXP), 256, 0, stream>>>(
        hbuf, W2t, b2, perm, cnt, ybuf);

    moe_combine<<<(N_TOK * DIM / 4 + 255) / 256, 256, 0, stream>>>(ybuf, gw, out);
}
